// T5Attention_12610023981528
// MI455X (gfx1250) — matrix operand order repaired and checked by reading, not compiled, not run on hardware
//
#include <hip/hip_runtime.h>
#include <stdint.h>

#define B_  2
#define S_  2048
#define D_  1024
#define H_  16
#define DK_ 64

typedef unsigned short u16;
typedef __attribute__((ext_vector_type(16))) __bf16 bf16x16;
typedef __attribute__((ext_vector_type(8)))  float  f32x8;

// ---------- helpers ----------

__device__ __forceinline__ u16 f2bf(float f) {
  uint32_t u = __float_as_uint(f);
  u += 0x7FFFu + ((u >> 16) & 1u);       // round-to-nearest-even
  return (u16)(u >> 16);
}

// A-fragment (16x32 bf16, M x K): lane r(0..15) = row, half selects K sub-ranges.
__device__ __forceinline__ bf16x16 load_a(const u16* src, int ld, int row0, int k0) {
  int lane = threadIdx.x & 31;
  int r = lane & 15, hf = lane >> 4;
  const u16* p = src + (size_t)(row0 + r) * ld + k0 + 8 * hf;
  bf16x16 a;
  ((uint4*)&a)[0] = *(const uint4*)p;
  ((uint4*)&a)[1] = *(const uint4*)(p + 16);
  return a;
}

// B-fragment (32x16 bf16, K x N) from an N-major (transposed) source:
// lane n(0..15) = column; half0 holds K 0..7,16..23, half1 K 8..15,24..31.
__device__ __forceinline__ bf16x16 load_bt(const u16* srcT, int ld, int n0, int k0) {
  int lane = threadIdx.x & 31;
  int n = lane & 15, hf = lane >> 4;
  const u16* p = srcT + (size_t)(n0 + n) * ld + k0 + 8 * hf;
  bf16x16 b;
  ((uint4*)&b)[0] = *(const uint4*)p;
  ((uint4*)&b)[1] = *(const uint4*)(p + 16);
  return b;
}

__device__ __forceinline__ f32x8 wmma_bf16(bf16x16 a, bf16x16 b, f32x8 c) {
  return __builtin_amdgcn_wmma_f32_16x16x32_bf16(false, a, false, b, (short)0, c, false, false);
}

// Async HBM -> LDS copy of one 16-byte chunk (gfx1250 ASYNCcnt path).
__device__ __forceinline__ void async_b128(uint32_t ldsoff, const void* gaddr) {
  asm volatile("global_load_async_to_lds_b128 %0, %1, off"
               :: "v"(ldsoff), "v"(gaddr)
               : "memory");
}

__device__ __forceinline__ uint32_t lds_off(const void* p) {
  // low 32 bits of a generic shared pointer = LDS byte offset
  return (uint32_t)(uintptr_t)p;
}

// T5 bidirectional relative-position bucket (num_buckets=32, max_distance=128)
__device__ __forceinline__ int rp_bucket(int rel) {
  int ret = (rel > 0) ? 16 : 0;
  int rp = (rel < 0) ? -rel : rel;
  int val;
  if (rp < 8) {
    val = rp;
  } else {
    float l = __logf((float)rp * 0.125f) * (8.0f / __logf(16.0f));
    int large = 8 + (int)l;
    val = large < 15 ? large : 15;
  }
  return ret + val;
}

// ---------- kernels ----------

__global__ void convert_f32_bf16(const float* __restrict__ x, u16* __restrict__ y, int n) {
  int i = blockIdx.x * blockDim.x + threadIdx.x;
  if (i < n) y[i] = f2bf(x[i]);
}

// Convert + transpose the 4 weight matrices: Wt[n][k] = bf16(W[k][n])
__global__ void convert_w_kernel(const float* __restrict__ Wq, const float* __restrict__ Wk,
                                 const float* __restrict__ Wv, const float* __restrict__ Wo,
                                 u16* __restrict__ Wqt, u16* __restrict__ Wkt,
                                 u16* __restrict__ Wvt, u16* __restrict__ Wot) {
  int z = blockIdx.y;
  const float* W = (z == 0) ? Wq : (z == 1) ? Wk : (z == 2) ? Wv : Wo;
  u16* Wt = (z == 0) ? Wqt : (z == 1) ? Wkt : (z == 2) ? Wvt : Wot;
  int i = blockIdx.x * blockDim.x + threadIdx.x;
  if (i < D_ * D_) {
    int k = i >> 10, n = i & 1023;
    Wt[(size_t)n * D_ + k] = f2bf(W[i]);
  }
}

// Per-head 1-D bias LUTs: b1[h][k-q+2047], b2[h][dq32+31], b3[h][dq64+31]
__global__ void bias_lut_kernel(const float* __restrict__ rel_bias,
                                const float* __restrict__ h_bias,
                                const float* __restrict__ v_bias,
                                float* __restrict__ b1, float* __restrict__ b2,
                                float* __restrict__ b3) {
  int h = blockIdx.x;
  for (int i = threadIdx.x; i < 2 * S_ - 1; i += blockDim.x) {
    int rel = i - (S_ - 1);
    b1[h * (2 * S_ - 1) + i] = rel_bias[rp_bucket(rel) * H_ + h];
  }
  for (int i = threadIdx.x; i < 63; i += blockDim.x) {
    int rel = i - 31;
    b2[h * 64 + i] = h_bias[rp_bucket(rel) * H_ + h];
    b3[h * 64 + i] = v_bias[rp_bucket(rel) * H_ + h];
  }
}

// QKV projections: C[4096 x 1024] = Xbf * W, 64x64 tile per workgroup (4 waves).
__global__ void __launch_bounds__(128)
qkv_gemm_kernel(const u16* __restrict__ Xbf,
                const u16* __restrict__ Wqt, const u16* __restrict__ Wkt,
                const u16* __restrict__ Wvt,
                u16* __restrict__ Qb, u16* __restrict__ Kb, u16* __restrict__ Vtb) {
  int z = blockIdx.z;
  const u16* Wt = (z == 0) ? Wqt : (z == 1) ? Wkt : Wvt;
  int wave = threadIdx.x >> 5, lane = threadIdx.x & 31;
  int m0 = blockIdx.y * 64 + wave * 16;
  int n0 = blockIdx.x * 64;
  f32x8 acc[4] = {};
  for (int k0 = 0; k0 < D_; k0 += 32) {
    bf16x16 a = load_a(Xbf, D_, m0, k0);
#pragma unroll
    for (int nb = 0; nb < 4; ++nb)
      acc[nb] = wmma_bf16(a, load_bt(Wt, D_, n0 + nb * 16, k0), acc[nb]);
  }
  int n = lane & 15, hf = lane >> 4;
#pragma unroll
  for (int nb = 0; nb < 4; ++nb) {
    int col = n0 + nb * 16 + n;
    int hh = col >> 6, dd = col & 63;
#pragma unroll
    for (int r = 0; r < 8; ++r) {
      int m = m0 + r + 8 * hf;
      int bb = m >> 11, ss = m & 2047;
      u16 val = f2bf(acc[nb][r]);
      if (z == 2)
        Vtb[(((size_t)bb * H_ + hh) * DK_ + dd) * S_ + ss] = val;
      else if (z == 0)
        Qb[(((size_t)bb * H_ + hh) * S_ + ss) * DK_ + dd] = val;
      else
        Kb[(((size_t)bb * H_ + hh) * S_ + ss) * DK_ + dd] = val;
    }
  }
}

// Flash attention: workgroup = (b,h,q-tile of 64), wave owns 16 q rows.
// K/V tiles staged through LDS with double-buffered async copies (ASYNCcnt).
__global__ void __launch_bounds__(128)
attn_kernel(const u16* __restrict__ Qb, const u16* __restrict__ Kb,
            const u16* __restrict__ Vtb,
            const float* __restrict__ b1, const float* __restrict__ b2,
            const float* __restrict__ b3, u16* __restrict__ Obuf) {
  __shared__ float s_b1[2 * S_ - 1];
  __shared__ float s_b2[64], s_b3[64];
  __shared__ __align__(16) u16 s_P[4][16 * 64];
  __shared__ __align__(16) u16 s_K[2][64 * 64];   // [key pos][dk]
  __shared__ __align__(16) u16 s_V[2][64 * 64];   // [dk][key pos local]

  int b = blockIdx.z, h = blockIdx.y;
  int wave = threadIdx.x >> 5, lane = threadIdx.x & 31;
  int n = lane & 15, hf = lane >> 4;
  int q0 = blockIdx.x * 64 + wave * 16;

  for (int i = threadIdx.x; i < 2 * S_ - 1; i += 128) s_b1[i] = b1[h * (2 * S_ - 1) + i];
  for (int i = threadIdx.x; i < 63; i += 128) {
    s_b2[i] = b2[h * 64 + i];
    s_b3[i] = b3[h * 64 + i];
  }

  const u16* Qp = Qb + (size_t)(b * H_ + h) * S_ * DK_;
  const u16* Kp = Kb + (size_t)(b * H_ + h) * S_ * DK_;
  const u16* Vp = Vtb + (size_t)(b * H_ + h) * DK_ * S_;

  bf16x16 qa0 = load_a(Qp, DK_, q0, 0);
  bf16x16 qa1 = load_a(Qp, DK_, q0, 32);

  float m[8], l[8];
  f32x8 o[4] = {};
#pragma unroll
  for (int r = 0; r < 8; ++r) { m[r] = -1e30f; l[r] = 0.f; }

  u16* myP = &s_P[wave][0];

  // --- async tile copy: 8 KB K tile (contiguous) + 8 KB V tile (64 strided rows)
  auto issue_tile = [&](int tile, int buf) {
    const char* gK = (const char*)(Kp + (size_t)tile * 64 * DK_);   // contiguous 8192 B
    uint32_t lK = lds_off(&s_K[buf][0]);
    uint32_t lV = lds_off(&s_V[buf][0]);
    const u16* gV = Vp + tile * 64;                                 // row stride S_
#pragma unroll
    for (int j = 0; j < 4; ++j) {
      int c = threadIdx.x + j * 128;          // chunk id 0..511
      async_b128(lK + c * 16, gK + c * 16);
      int d = c >> 3, cc = c & 7;
      async_b128(lV + c * 16, (const char*)(gV + (size_t)d * S_) + cc * 16);
    }
  };

  const int NT = S_ / 64;
  issue_tile(0, 0);

  for (int it = 0; it < NT; ++it) {
    int buf = it & 1;
    int kt = it * 64;
    if (it + 1 < NT) {
      issue_tile(it + 1, buf ^ 1);
      asm volatile("s_wait_asynccnt 0x8" ::: "memory");
    } else {
      asm volatile("s_wait_asynccnt 0x0" ::: "memory");
    }
    __syncthreads();   // tile `it` visible to all waves (also covers bias LUT on it==0)

    const u16* Ktile = &s_K[buf][0];
    const u16* Vtile = &s_V[buf][0];

    // scores: 16x64 = Q(16x64) * K(64x64)^T   (B-fragments from LDS)
    f32x8 sc[4];
#pragma unroll
    for (int nb = 0; nb < 4; ++nb) {
      f32x8 c = {};
      c = wmma_bf16(qa0, load_bt(Ktile, DK_, nb * 16, 0), c);
      c = wmma_bf16(qa1, load_bt(Ktile, DK_, nb * 16, 32), c);
      sc[nb] = c;
    }
    // add relative-position bias; track row max
    float rmax[8];
#pragma unroll
    for (int r = 0; r < 8; ++r) rmax[r] = -1e30f;
#pragma unroll
    for (int nb = 0; nb < 4; ++nb) {
      int k = kt + nb * 16 + n;
#pragma unroll
      for (int r = 0; r < 8; ++r) {
        int q = q0 + r + 8 * hf;
        float bias = s_b1[k - q + (S_ - 1)] +
                     s_b2[(q & 31) - (k & 31) + 31] +
                     s_b3[(q >> 6) - (k >> 6) + 31];
        float v = sc[nb][r] + bias;
        sc[nb][r] = v;
        rmax[r] = fmaxf(rmax[r], v);
      }
    }
#pragma unroll
    for (int r = 0; r < 8; ++r) {
#pragma unroll
      for (int msk = 1; msk < 16; msk <<= 1)
        rmax[r] = fmaxf(rmax[r], __shfl_xor(rmax[r], msk, 32));
    }
    float scale[8], rsum[8];
#pragma unroll
    for (int r = 0; r < 8; ++r) {
      float mn = fmaxf(m[r], rmax[r]);
      scale[r] = __expf(m[r] - mn);
      m[r] = mn;
      rsum[r] = 0.f;
    }
#pragma unroll
    for (int nb = 0; nb < 4; ++nb) {
#pragma unroll
      for (int r = 0; r < 8; ++r) {
        float p = __expf(sc[nb][r] - m[r]);
        sc[nb][r] = p;
        rsum[r] += p;
      }
    }
#pragma unroll
    for (int r = 0; r < 8; ++r) {
#pragma unroll
      for (int msk = 1; msk < 16; msk <<= 1)
        rsum[r] += __shfl_xor(rsum[r], msk, 32);
      l[r] = l[r] * scale[r] + rsum[r];
    }
#pragma unroll
    for (int nb = 0; nb < 4; ++nb)
#pragma unroll
      for (int r = 0; r < 8; ++r) o[nb][r] *= scale[r];

    // stage P (C-layout f32 -> row-major bf16 in LDS), wave-private
#pragma unroll
    for (int nb = 0; nb < 4; ++nb)
#pragma unroll
      for (int r = 0; r < 8; ++r)
        myP[(r + 8 * hf) * 64 + nb * 16 + n] = f2bf(sc[nb][r]);
    asm volatile("s_wait_dscnt 0x0" ::: "memory");

    // O += P(16x64) * V(64x64), V tile already transposed in LDS
    bf16x16 pa0 = load_a(myP, 64, 0, 0);
    bf16x16 pa1 = load_a(myP, 64, 0, 32);
#pragma unroll
    for (int nb = 0; nb < 4; ++nb) {
      o[nb] = wmma_bf16(pa0, load_bt(Vtile, 64, nb * 16, 0), o[nb]);
      o[nb] = wmma_bf16(pa1, load_bt(Vtile, 64, nb * 16, 32), o[nb]);
    }
    __syncthreads();   // everyone done reading buf before it gets overwritten
  }

  // normalize and write O [B,S,H*DK] bf16
#pragma unroll
  for (int nb = 0; nb < 4; ++nb) {
    int col = h * 64 + nb * 16 + n;
#pragma unroll
    for (int r = 0; r < 8; ++r) {
      int q = q0 + r + 8 * hf;
      Obuf[((size_t)b * S_ + q) * (H_ * DK_) + col] = f2bf(o[nb][r] / l[r]);
    }
  }
}

// Output projection: Y[4096x1024] f32 = Obuf(bf16) * Wo
__global__ void __launch_bounds__(128)
out_gemm_kernel(const u16* __restrict__ Obuf, const u16* __restrict__ Wot,
                float* __restrict__ Y) {
  int wave = threadIdx.x >> 5, lane = threadIdx.x & 31;
  int m0 = blockIdx.y * 64 + wave * 16;
  int n0 = blockIdx.x * 64;
  f32x8 acc[4] = {};
  for (int k0 = 0; k0 < D_; k0 += 32) {
    bf16x16 a = load_a(Obuf, D_, m0, k0);
#pragma unroll
    for (int nb = 0; nb < 4; ++nb)
      acc[nb] = wmma_bf16(a, load_bt(Wot, D_, n0 + nb * 16, k0), acc[nb]);
  }
  int n = lane & 15, hf = lane >> 4;
#pragma unroll
  for (int nb = 0; nb < 4; ++nb)
#pragma unroll
    for (int r = 0; r < 8; ++r)
      Y[(size_t)(m0 + r + 8 * hf) * D_ + n0 + nb * 16 + n] = acc[nb][r];
}

// ---------- host ----------

extern "C" void kernel_launch(void* const* d_in, const int* in_sizes, int n_in,
                              void* d_out, int out_size, void* d_ws, size_t ws_size,
                              hipStream_t stream) {
  (void)in_sizes; (void)n_in; (void)out_size; (void)ws_size;
  const float* X  = (const float*)d_in[0];
  const float* Wq = (const float*)d_in[1];
  const float* Wk = (const float*)d_in[2];
  const float* Wv = (const float*)d_in[3];
  const float* Wo = (const float*)d_in[4];
  const float* rel_bias = (const float*)d_in[5];
  const float* h_bias   = (const float*)d_in[6];
  const float* v_bias   = (const float*)d_in[7];
  float* out = (float*)d_out;

  char* w = (char*)d_ws;
  size_t off = 0;
  auto alloc_u16 = [&](size_t nelem) { u16* p = (u16*)(w + off); off += nelem * 2; return p; };
  u16* Xbf = alloc_u16((size_t)B_ * S_ * D_);
  u16* Wqt = alloc_u16((size_t)D_ * D_);
  u16* Wkt = alloc_u16((size_t)D_ * D_);
  u16* Wvt = alloc_u16((size_t)D_ * D_);
  u16* Wot = alloc_u16((size_t)D_ * D_);
  u16* Qb  = alloc_u16((size_t)B_ * H_ * S_ * DK_);
  u16* Kb  = alloc_u16((size_t)B_ * H_ * S_ * DK_);
  u16* Vtb = alloc_u16((size_t)B_ * H_ * DK_ * S_);
  u16* Obuf = alloc_u16((size_t)B_ * S_ * H_ * DK_);
  off = (off + 15) & ~(size_t)15;
  float* b1 = (float*)(w + off); off += (size_t)H_ * (2 * S_ - 1) * 4;
  float* b2 = (float*)(w + off); off += (size_t)H_ * 64 * 4;
  float* b3 = (float*)(w + off); off += (size_t)H_ * 64 * 4;

  int nX = B_ * S_ * D_;
  convert_f32_bf16<<<(nX + 255) / 256, 256, 0, stream>>>(X, Xbf, nX);
  convert_w_kernel<<<dim3((D_ * D_ + 255) / 256, 4), 256, 0, stream>>>(
      Wq, Wk, Wv, Wo, Wqt, Wkt, Wvt, Wot);
  bias_lut_kernel<<<H_, 256, 0, stream>>>(rel_bias, h_bias, v_bias, b1, b2, b3);
  qkv_gemm_kernel<<<dim3(D_ / 64, (B_ * S_) / 64, 3), 128, 0, stream>>>(
      Xbf, Wqt, Wkt, Wvt, Qb, Kb, Vtb);
  attn_kernel<<<dim3(S_ / 64, H_, B_), 128, 0, stream>>>(Qb, Kb, Vtb, b1, b2, b3, Obuf);
  out_gemm_kernel<<<dim3(D_ / 64, (B_ * S_) / 64), 128, 0, stream>>>(Obuf, Wot, out);
}
